// GAT_layer_28595892257581
// MI455X (gfx1250) — matrix operand order repaired and checked by reading, not compiled, not run on hardware
//
#include <hip/hip_runtime.h>
#include <hip/hip_bf16.h>

#define HEADS 8
#define DIN   256
#define HD    64
#define NBAT  4
#define NN    2048
#define BHP   (NBAT*HEADS)   // 32 (batch,head) pairs

typedef __attribute__((ext_vector_type(16))) __bf16 v16bf;
typedef __attribute__((ext_vector_type(8)))  float  v8f;

union FragAB { unsigned int u[8]; v16bf v; };

__device__ __forceinline__ unsigned int f2bf1(float f) {
    unsigned int u = __float_as_uint(f);
    return (u + 0x7FFFu + ((u >> 16) & 1u)) >> 16;   // RNE truncate to bf16
}
__device__ __forceinline__ unsigned int pack2bf(float a, float b) {
    return f2bf1(a) | (f2bf1(b) << 16);
}

// ---------------------------------------------------------------------------
// Kernel 1: h_[b,h] = h[b] @ w[h]  (2048x256 @ 256x64, bf16 WMMA, f32 acc)
// Software-pipelined: global loads for K-step i+1 issue right after the
// barrier and their wait lands only at the LDS-store at the top of the next
// iteration -> latency hidden behind the WMMA/VALU compute phase.
// Stores h_ transposed as bf16  hT[bh][o][n]  and src/dst logit vectors.
// Block = 256 thr (8 waves); wave -> 16 rows x 64 cols. Grid (16 iblk, 32 bh).
// ---------------------------------------------------------------------------
__global__ __launch_bounds__(256) void gat_proj(
    const float* __restrict__ h, const float* __restrict__ w,
    const float* __restrict__ a_src, const float* __restrict__ a_dst,
    unsigned short* __restrict__ hT, float* __restrict__ srcw,
    float* __restrict__ dstw)
{
    __shared__ unsigned short ldsA[2][128 * 34];   // 128 rows x 32 k, pad->34
    __shared__ unsigned short ldsB[2][64 * 34];    // 64 o  x 32 k (transposed w)

    const int iblk = blockIdx.x, bh = blockIdx.y;
    const int bi = bh >> 3, hd = bh & 7;
    const int t = threadIdx.x;
    const int wid = t >> 5, lane = t & 31;
    const int lo = lane & 15, hi = lane >> 4;

    v8f c[4];
#pragma unroll
    for (int nt = 0; nt < 4; ++nt)
#pragma unroll
        for (int v = 0; v < 8; ++v) c[nt][v] = 0.0f;

    const float* hbase = h + ((size_t)bi * NN + (size_t)iblk * 128) * DIN;
    const int arowA = t >> 1, aks = (t & 1) * 16;  // staging coords (A)
    const int bk = t >> 3, bob = (t & 7) * 8;      // staging coords (B)

    float ga[16], gb[8];                           // in-flight global data
    auto loadG = [&](int kb) {
        const float* g = hbase + (size_t)arowA * DIN + kb + aks;
#pragma unroll
        for (int i = 0; i < 16; ++i) ga[i] = g[i];
        const float* gw = w + ((size_t)hd * DIN + kb + bk) * HD + bob;
#pragma unroll
        for (int i = 0; i < 8; ++i) gb[i] = gw[i];
    };
    auto storeL = [&](int buf) {
        unsigned int* dp = (unsigned int*)&ldsA[buf][arowA * 34 + aks];
#pragma unroll
        for (int i = 0; i < 8; ++i) dp[i] = pack2bf(ga[2 * i], ga[2 * i + 1]);
#pragma unroll
        for (int i = 0; i < 8; ++i)
            ldsB[buf][(bob + i) * 34 + bk] = (unsigned short)f2bf1(gb[i]);
    };

    loadG(0);
    for (int it = 0; it < DIN / 32; ++it) {
        const int buf = it & 1;
        storeL(buf);                               // waits the global loads here
        __syncthreads();                           // includes s_wait_dscnt 0
        if (it + 1 < DIN / 32) loadG((it + 1) * 32);   // overlap with compute

        // gather ALL fragments first (one ds clause, one wait), then 4 WMMAs
        FragAB bfr[4];
#pragma unroll
        for (int nt = 0; nt < 4; ++nt) {
            const int o = nt * 16 + lo;            // lane = N (col)
#pragma unroll
            for (int p = 0; p < 8; ++p)            // elem pair -> K = 2p + 16*hi
                bfr[nt].u[p] = *(const unsigned int*)&ldsB[buf][o * 34 + 2 * p + 16 * hi];
        }
        FragAB a;                                  // lane = M, VGPR p -> K map
        const int arow = wid * 16 + lo;
#pragma unroll
        for (int p = 0; p < 8; ++p) {
            const int k = 2 * p + 16 * hi;
            a.u[p] = *(const unsigned int*)&ldsA[buf][arow * 34 + k];
        }
#pragma unroll
        for (int nt = 0; nt < 4; ++nt)
            c[nt] = __builtin_amdgcn_wmma_f32_16x16x32_bf16(
                false, a.v, false, bfr[nt].v, (short)0, c[nt], false, false);
    }

    // epilogue: store h_ transposed (bf16) + src/dst logits from f32 acc
    const int rowbase = iblk * 128 + wid * 16;
    float sa[8], sd[8];
#pragma unroll
    for (int v = 0; v < 8; ++v) { sa[v] = 0.0f; sd[v] = 0.0f; }

#pragma unroll
    for (int nt = 0; nt < 4; ++nt) {
        const int o = nt * 16 + lo;
        const float asv = a_src[hd * HD + o];
        const float adv = a_dst[hd * HD + o];
        const size_t obase = ((size_t)bh * HD + o) * NN + rowbase;
#pragma unroll
        for (int v = 0; v < 8; ++v) {
            const float val = c[nt][v];
            hT[obase + v + 8 * hi] = (unsigned short)f2bf1(val);
            sa[v] += val * asv;
            sd[v] += val * adv;
        }
    }
    // reduce over the 16 lanes of each half (rows differ only across halves)
#pragma unroll
    for (int m = 1; m < 16; m <<= 1)
#pragma unroll
        for (int v = 0; v < 8; ++v) {
            sa[v] += __shfl_xor(sa[v], m, 32);
            sd[v] += __shfl_xor(sd[v], m, 32);
        }
    if (lo == 0) {
#pragma unroll
        for (int v = 0; v < 8; ++v) {
            const int r = bh * NN + rowbase + v + 8 * hi;
            srcw[r] = sa[v];
            dstw[r] = sd[v];
        }
    }
}

// ---------------------------------------------------------------------------
// Kernel 2: per-row softmax stats: M = max_j lrelu(src_i+dst_j),
// Linv = 1/sum_j exp(.-M). One wave per row; dst vector staged once in LDS
// (all 8 rows of a block share the same (b,h) -> same dst[2048]).
// ---------------------------------------------------------------------------
__global__ __launch_bounds__(256) void gat_stats(
    const float* __restrict__ srcw, const float* __restrict__ dstw,
    float* __restrict__ Mw, float* __restrict__ Lw)
{
    __shared__ float dls[NN];                      // 8 KB
    const int t = threadIdx.x, wid = t >> 5, lane = t & 31;
    const int rg = blockIdx.x * 8 + wid;           // global row
    const int bh = rg >> 11;                       // 2048 % 8 == 0: block-uniform

    for (int j = t; j < NN; j += 256)
        dls[j] = dstw[(size_t)bh * NN + j];
    __syncthreads();

    const float sv = srcw[rg];

    float m = -3.0e38f;
    for (int j = lane; j < NN; j += 32) {
        float x = sv + dls[j];
        x = x > 0.0f ? x : 0.2f * x;
        m = fmaxf(m, x);
    }
#pragma unroll
    for (int msk = 16; msk >= 1; msk >>= 1)
        m = fmaxf(m, __shfl_xor(m, msk, 32));

    float s = 0.0f;
    for (int j = lane; j < NN; j += 32) {
        float x = sv + dls[j];
        x = x > 0.0f ? x : 0.2f * x;
        s += __expf(x - m);
    }
#pragma unroll
    for (int msk = 16; msk >= 1; msk >>= 1)
        s += __shfl_xor(s, msk, 32);

    if (lane == 0) { Mw[rg] = m; Lw[rg] = 1.0f / s; }
}

// ---------------------------------------------------------------------------
// Kernel 3: h_out = softmax(P) @ h_ streamed over j-tiles of 32 (no NxN mat).
// Software-pipelined like kernel 1; B-fragments gathered as one ds clause
// (overlapping the exp/pack VALU chain), then 4 hazard-free WMMAs.
// Block = 256 thr (8 waves); wave -> 16 rows x 64 cols. Grid (16 iblk, 32 bh).
// ---------------------------------------------------------------------------
__global__ __launch_bounds__(256) void gat_attn(
    const unsigned short* __restrict__ hT, const float* __restrict__ srcw,
    const float* __restrict__ dstw, const float* __restrict__ Mw,
    const float* __restrict__ Lw, const float* __restrict__ bias,
    float* __restrict__ out)
{
    __shared__ unsigned short ldsV[2][64 * 34];    // V tile: [o][j], bf16
    __shared__ float dstS[2][32];

    const int iblk = blockIdx.x, bh = blockIdx.y;
    const int t = threadIdx.x;
    const int wid = t >> 5, lane = t & 31;
    const int lo = lane & 15, hi = lane >> 4;
    const int rowbase = iblk * 128 + wid * 16;

    const int r = bh * NN + rowbase + lo;          // this lane's P-row stats
    const float srow = srcw[r];
    const float mrow = Mw[r];
    const float lrow = Lw[r];

    v8f c[4];
#pragma unroll
    for (int nt = 0; nt < 4; ++nt)
#pragma unroll
        for (int v = 0; v < 8; ++v) c[nt][v] = 0.0f;

    const unsigned short* vbase = hT + (size_t)bh * HD * NN;
    const int so = t >> 2, sko = (t & 3) * 8;      // staging coords

    uint4 gv;                                      // in-flight global data
    float gd = 0.0f;
    auto loadG = [&](int jb) {
        gv = *(const uint4*)(vbase + (size_t)so * NN + jb + sko);
        if (t < 32) gd = dstw[(size_t)bh * NN + jb + t];
    };
    auto storeL = [&](int buf) {
        unsigned int* ls = (unsigned int*)&ldsV[buf][so * 34 + sko];
        ls[0] = gv.x; ls[1] = gv.y; ls[2] = gv.z; ls[3] = gv.w;
        if (t < 32) dstS[buf][t] = gd;
    };

    loadG(0);
    for (int it = 0; it < NN / 32; ++it) {
        const int buf = it & 1;
        storeL(buf);                               // waits the global loads here
        __syncthreads();                           // includes s_wait_dscnt 0
        if (it + 1 < NN / 32) loadG((it + 1) * 32);    // overlap with compute

        // issue all B-fragment LDS loads first ...
        FragAB bfr[4];
#pragma unroll
        for (int nt = 0; nt < 4; ++nt) {
            const int o = nt * 16 + lo;
#pragma unroll
            for (int p = 0; p < 8; ++p)
                bfr[nt].u[p] = *(const unsigned int*)&ldsV[buf][o * 34 + 2 * p + 16 * hi];
        }
        // ... overlapped with the exp/pack VALU chain building the A fragment
        FragAB a;
#pragma unroll
        for (int p = 0; p < 8; ++p) {
            const int k0 = 2 * p + 16 * hi;
            float x0 = srow + dstS[buf][k0];
            float x1 = srow + dstS[buf][k0 + 1];
            x0 = x0 > 0.0f ? x0 : 0.2f * x0;
            x1 = x1 > 0.0f ? x1 : 0.2f * x1;
            const float p0 = __expf(x0 - mrow) * lrow;
            const float p1 = __expf(x1 - mrow) * lrow;
            a.u[p] = pack2bf(p0, p1);
        }
#pragma unroll
        for (int nt = 0; nt < 4; ++nt)
            c[nt] = __builtin_amdgcn_wmma_f32_16x16x32_bf16(
                false, a.v, false, bfr[nt].v, (short)0, c[nt], false, false);
    }

    // coalesced epilogue: out[bh][row][o] = acc + bias[o]
#pragma unroll
    for (int nt = 0; nt < 4; ++nt) {
        const int o = nt * 16 + lo;
        const float bb = bias[o];
#pragma unroll
        for (int v = 0; v < 8; ++v) {
            const int row = rowbase + v + 8 * hi;
            out[((size_t)bh * NN + row) * HD + o] = c[nt][v] + bb;
        }
    }
}

// ---------------------------------------------------------------------------
extern "C" void kernel_launch(void* const* d_in, const int* in_sizes, int n_in,
                              void* d_out, int out_size, void* d_ws, size_t ws_size,
                              hipStream_t stream)
{
    (void)in_sizes; (void)n_in; (void)out_size; (void)ws_size;
    const float* h     = (const float*)d_in[0];
    const float* w     = (const float*)d_in[1];
    const float* a_src = (const float*)d_in[2];
    const float* a_dst = (const float*)d_in[3];
    const float* bias  = (const float*)d_in[4];
    float* out = (float*)d_out;

    char* ws = (char*)d_ws;
    const size_t hT_bytes  = (size_t)BHP * HD * NN * 2;   // 8 MB bf16
    const size_t vec_bytes = (size_t)BHP * NN * 4;        // 256 KB each
    unsigned short* hT = (unsigned short*)ws;
    float* srcw = (float*)(ws + hT_bytes);
    float* dstw = (float*)(ws + hT_bytes + 1 * vec_bytes);
    float* Mw   = (float*)(ws + hT_bytes + 2 * vec_bytes);
    float* Lw   = (float*)(ws + hT_bytes + 3 * vec_bytes);

    dim3 grid(NN / 128, BHP);   // (16, 32)
    gat_proj <<<grid, 256, 0, stream>>>(h, w, a_src, a_dst, hT, srcw, dstw);
    gat_stats<<<(BHP * NN) / 8, 256, 0, stream>>>(srcw, dstw, Mw, Lw);
    gat_attn <<<grid, 256, 0, stream>>>(hT, srcw, dstw, Mw, Lw, bias, out);
}
